// SelfAttention_1348619731265
// MI455X (gfx1250) — matrix operand order repaired and checked by reading, not compiled, not run on hardware
//
#include <hip/hip_runtime.h>
#include <hip/hip_bf16.h>

// ---------- types ----------
typedef _Float16 h16 __attribute__((ext_vector_type(16)));
typedef _Float16 h8  __attribute__((ext_vector_type(8)));
typedef float    v8f __attribute__((ext_vector_type(8)));
typedef float    f8  __attribute__((ext_vector_type(8)));

// problem constants (from reference): B=4, H=W=64 -> N=4096 tokens/batch, C=F=128
#define NB    4
#define NTOK  4096
#define CH    128
#define MT    (NB*NTOK)          // 16384 total tokens
#define NSEG  2                  // split-K segments (flash-decoding)
#define SEGK  (NTOK/NSEG)        // 2048 keys per segment
#define NQB   (MT/16)            // 1024 query blocks (16 rows each)

// cross-lane xor "shuffle" within the wave via ds_swizzle (and=0x1f, or=0, xor=MASK)
#define XORSWZ(x, MASK) \
    __uint_as_float((unsigned)__builtin_amdgcn_ds_swizzle((int)__float_as_uint(x), (((MASK) << 10) | 0x1f)))

// ---------- WMMA wrapper ----------
__device__ __forceinline__ v8f wmma_f16(h16 a, h16 b, v8f c) {
    // (neg_a, A, neg_b, B, c_mod, C, reuse_a, reuse_b)
    return __builtin_amdgcn_wmma_f32_16x16x32_f16(false, a, false, b, (short)0, c, false, false);
}

// ---------- fragment loaders ----------
// A-matrix 16x32 f16: lane = m + 16*h; elem e: K = h*16 + e
__device__ __forceinline__ h16 load_a_f16(const _Float16* base, int ld, int row, int c0, int hi) {
    const _Float16* p = base + (size_t)row * ld + c0 + hi * 16;
    h8 r0 = *(const h8*)p;
    h8 r1 = *(const h8*)(p + 8);
    h16 a;
#pragma unroll
    for (int i = 0; i < 8; ++i) { a[i] = r0[i]; a[8 + i] = r1[i]; }
    return a;
}
// same A layout but source is f32 (convert to f16)
__device__ __forceinline__ h16 load_a_f32(const float* base, int ld, int row, int c0, int hi) {
    const float* p = base + (size_t)row * ld + c0 + hi * 16;
    f8 r0 = *(const f8*)p;
    f8 r1 = *(const f8*)(p + 8);
    h16 a;
#pragma unroll
    for (int i = 0; i < 8; ++i) { a[i] = (_Float16)r0[i]; a[8 + i] = (_Float16)r1[i]; }
    return a;
}
// B-matrix 32x16 f16: lane = n + 16*h; elem e: K = h*16 + e  -> 16 contiguous f16 per lane
__device__ __forceinline__ h16 load_b_f16(const _Float16* base, int ld, int row, int c0, int hi) {
    return *(const h16*)(base + (size_t)row * ld + c0 + hi * 16);
}

// ---------- kernel 0: transpose+convert weights: Wt[p][f][c] = W_p[c][f] (f32->f16) ----------
__global__ void wconv_kernel(const float* __restrict__ Wq, const float* __restrict__ Wk,
                             const float* __restrict__ Wv, _Float16* __restrict__ Wt) {
    int i = blockIdx.x * blockDim.x + threadIdx.x;
    if (i >= 3 * CH * CH) return;
    int p = i >> 14;            // /16384
    int rem = i & 16383;
    int c = rem >> 7;           // /128
    int f = rem & 127;
    const float* W = (p == 0) ? Wq : (p == 1) ? Wk : Wv;
    Wt[(size_t)p * CH * CH + (size_t)f * CH + c] = (_Float16)W[(size_t)c * CH + f];
}

// ---------- kernel 1: projections Q,K (row-major f16) and V (transposed f16 Vt[b][f][tok]) ----------
__global__ __launch_bounds__(32)
void proj_kernel(const float* __restrict__ x, const _Float16* __restrict__ Wt,
                 const float* __restrict__ bq, const float* __restrict__ bk,
                 const float* __restrict__ bv,
                 _Float16* __restrict__ Q, _Float16* __restrict__ K, _Float16* __restrict__ Vt) {
    const int lane = threadIdx.x & 31;
    const int lo = lane & 15, hi = lane >> 4;
    const int mbase = blockIdx.x * 16;          // token tile
    const int gy = blockIdx.y;
    const int p = gy >> 3;                      // 0=Q 1=K 2=V
    const int fb = (gy & 7) * 16;               // feature tile base

    const _Float16* Wp = Wt + (size_t)p * CH * CH;

    v8f acc = {};
#pragma unroll
    for (int c = 0; c < 4; ++c) {
        h16 a = load_a_f32(x, CH, mbase + lo, 32 * c, hi);
        h16 b = load_b_f16(Wp, CH, fb + lo, 32 * c, hi);
        acc = wmma_f16(a, b, acc);
    }
    const float* bias = (p == 0) ? bq : (p == 1) ? bk : bv;
    const float bb = bias[fb + lo];             // column N = lo for all 8 rows of this lane
#pragma unroll
    for (int r = 0; r < 8; ++r) acc[r] += bb;

    if (p < 2) {
        _Float16* dst = (p == 0) ? Q : K;
#pragma unroll
        for (int r = 0; r < 8; ++r) {
            int tok = mbase + r + 8 * hi;       // D: M = r + 8*h
            dst[(size_t)tok * CH + fb + lo] = (_Float16)acc[r];
        }
    } else {
        // transposed store: Vt[b][f][tok]; per-lane 8 consecutive tokens -> one b128 store
        int b = mbase >> 12;                    // /4096 (tile never crosses batch: 4096%16==0)
        int tok0 = (mbase & (NTOK - 1)) + 8 * hi;
        h8 v;
#pragma unroll
        for (int r = 0; r < 8; ++r) v[r] = (_Float16)acc[r];
        *(h8*)(Vt + ((size_t)b * CH + fb + lo) * NTOK + tok0) = v;
    }
}

// ---------- kernel 2: flash attention (split-K), one wave per (16-query block, key segment) ----------
// Writes UNNORMALIZED partial O (f32) and per-row (m, l) stats; merge kernel combines segments.
__global__ __launch_bounds__(128)
void attn_kernel(const _Float16* __restrict__ Q, const _Float16* __restrict__ K,
                 const _Float16* __restrict__ Vt,
                 float* __restrict__ Opart, float* __restrict__ Mst, float* __restrict__ Lst) {
    __shared__ __align__(16) _Float16 plds[4][16][40];   // 32 cols + 8 pad (16B-aligned rows, spread banks)

    const int b    = blockIdx.z;
    const int seg  = blockIdx.y;
    const int wave = threadIdx.x >> 5;
    const int lane = threadIdx.x & 31;
    const int lo = lane & 15, hi = lane >> 4;
    const int qblk  = blockIdx.x * 4 + wave;             // query block within batch [0,256)
    const int qbase = qblk * 16;
    const int qb    = b * (NTOK / 16) + qblk;            // global query block [0,1024)

    const _Float16* Qb = Q + (size_t)b * NTOK * CH;
    const _Float16* Kb = K + (size_t)b * NTOK * CH;
    const _Float16* Vb = Vt + (size_t)b * CH * NTOK;

    // Q A-fragments for the whole 128-wide contraction dim
    h16 qf[4];
#pragma unroll
    for (int c = 0; c < 4; ++c) qf[c] = load_a_f16(Qb, CH, qbase + lo, 32 * c, hi);

    // per-lane row stats for rows M = r + 8*hi (replicated across the 16 lanes of a half)
    float mrow[8], lrow[8];
    v8f o[8];
#pragma unroll
    for (int r = 0; r < 8; ++r) { mrow[r] = -__builtin_inff(); lrow[r] = 0.0f; }
#pragma unroll
    for (int ft = 0; ft < 8; ++ft) o[ft] = (v8f){};

    const int k0 = seg * SEGK, kend = k0 + SEGK;
    for (int kc = k0; kc < kend; kc += 32) {
        // prefetch next 32-key K tile (global_prefetch_b8); lanes cover 16 rows x 2 half-rows each
        __builtin_prefetch((const char*)(Kb + (size_t)(kc + 32 + lo) * CH + hi * 64), 0, 1);
        __builtin_prefetch((const char*)(Kb + (size_t)(kc + 48 + lo) * CH + hi * 64), 0, 1);

        // ---- S = Q K^T for two 16-key sub-tiles ----
        v8f s0 = {}, s1 = {};
#pragma unroll
        for (int c = 0; c < 4; ++c) {
            h16 kb0 = load_b_f16(Kb, CH, kc + lo, 32 * c, hi);
            s0 = wmma_f16(qf[c], kb0, s0);
        }
#pragma unroll
        for (int c = 0; c < 4; ++c) {
            h16 kb1 = load_b_f16(Kb, CH, kc + 16 + lo, 32 * c, hi);
            s1 = wmma_f16(qf[c], kb1, s1);
        }

        // ---- online softmax update ----
        float corr[8], p0[8], p1[8];
#pragma unroll
        for (int r = 0; r < 8; ++r) {
            float t = fmaxf(s0[r], s1[r]);
            t = fmaxf(t, XORSWZ(t, 1));
            t = fmaxf(t, XORSWZ(t, 2));
            t = fmaxf(t, XORSWZ(t, 4));
            t = fmaxf(t, XORSWZ(t, 8));          // row max over this 32-key chunk
            float nm = fmaxf(mrow[r], t);
            corr[r] = __expf(mrow[r] - nm);      // 0 on first chunk (-inf)
            mrow[r] = nm;
            p0[r] = __expf(s0[r] - nm);
            p1[r] = __expf(s1[r] - nm);
            float ps = p0[r] + p1[r];
            ps += XORSWZ(ps, 1);
            ps += XORSWZ(ps, 2);
            ps += XORSWZ(ps, 4);
            ps += XORSWZ(ps, 8);                 // row sum over chunk
            lrow[r] = lrow[r] * corr[r] + ps;
        }
        // rescale existing accumulators
#pragma unroll
        for (int ft = 0; ft < 8; ++ft)
#pragma unroll
            for (int r = 0; r < 8; ++r) o[ft][r] *= corr[r];

        // ---- P (f32 C-layout) -> LDS (f16) -> A-fragment ----
#pragma unroll
        for (int r = 0; r < 8; ++r) {
            plds[wave][r + 8 * hi][lo]      = (_Float16)p0[r];
            plds[wave][r + 8 * hi][16 + lo] = (_Float16)p1[r];
        }
        asm volatile("s_wait_dscnt 0" ::: "memory");
        const _Float16* pp = &plds[wave][lo][hi * 16];
        h8 a0 = *(const h8*)pp;
        h8 a1 = *(const h8*)(pp + 8);
        h16 pa;
#pragma unroll
        for (int i = 0; i < 8; ++i) { pa[i] = a0[i]; pa[8 + i] = a1[i]; }

        // ---- O += P * V  (B operand from transposed V: contiguous per lane) ----
#pragma unroll
        for (int ft = 0; ft < 8; ++ft) {
            h16 vb = *(const h16*)(Vb + ((size_t)(ft * 16 + lo)) * NTOK + kc + hi * 16);
            o[ft] = wmma_f16(pa, vb, o[ft]);
        }
    }

    // ---- write unnormalized partial O + stats ----
    float* obase = Opart + ((size_t)qb * NSEG + seg) * (16 * CH);
#pragma unroll
    for (int ft = 0; ft < 8; ++ft)
#pragma unroll
        for (int r = 0; r < 8; ++r)
            obase[(r + 8 * hi) * CH + ft * 16 + lo] = o[ft][r];
    if (lo == 0) {
        int sbase = (qb * NSEG + seg) * 16 + 8 * hi;
#pragma unroll
        for (int r = 0; r < 8; ++r) { Mst[sbase + r] = mrow[r]; Lst[sbase + r] = lrow[r]; }
    }
}

// ---------- kernel 3: merge split-K partials, normalize, add residual ----------
__global__ __launch_bounds__(256)
void merge_kernel(const float* __restrict__ x, const float* __restrict__ Opart,
                  const float* __restrict__ Mst, const float* __restrict__ Lst,
                  float* __restrict__ out) {
    int e = blockIdx.x * 256 + threadIdx.x;              // [0, MT*CH) ; e == output flat index
    int qb  = e >> 11;                                   // / (16*128)
    int rem = e & 2047;
    int row = rem >> 7;

    int s0i = (qb * NSEG + 0) * 16 + row;
    int s1i = (qb * NSEG + 1) * 16 + row;
    float m0 = Mst[s0i], m1 = Mst[s1i];
    float l0 = Lst[s0i], l1 = Lst[s1i];
    float ms = fmaxf(m0, m1);
    float w0 = __expf(m0 - ms), w1 = __expf(m1 - ms);
    float linv = 1.0f / (w0 * l0 + w1 * l1);

    float O0 = Opart[((size_t)qb * NSEG + 0) * 2048 + rem];
    float O1 = Opart[((size_t)qb * NSEG + 1) * 2048 + rem];
    out[e] = (w0 * O0 + w1 * O1) * linv + x[e];
}

// ---------- host launcher ----------
extern "C" void kernel_launch(void* const* d_in, const int* in_sizes, int n_in,
                              void* d_out, int out_size, void* d_ws, size_t ws_size,
                              hipStream_t stream) {
    const float* x  = (const float*)d_in[0];
    const float* Wq = (const float*)d_in[1];
    const float* bq = (const float*)d_in[2];
    const float* Wk = (const float*)d_in[3];
    const float* bk = (const float*)d_in[4];
    const float* Wv = (const float*)d_in[5];
    const float* bv = (const float*)d_in[6];
    float* out = (float*)d_out;

    // workspace layout:
    //  Wt  f16 [3][128][128]            96 KB   @ 0
    //  Q   f16 [16384][128]              4 MB   @ 98304
    //  K   f16 [16384][128]              4 MB   @ 4292608
    //  Vt  f16 [4][128][4096]            4 MB   @ 8486912
    //  Opart f32 [1024][2][16][128]     16 MB   @ 12681216
    //  Mst f32 [1024*2*16]             128 KB   @ 29458432
    //  Lst f32 [1024*2*16]             128 KB   @ 29589504
    char* ws = (char*)d_ws;
    _Float16* Wt  = (_Float16*)ws;
    _Float16* Qb  = (_Float16*)(ws + 98304);
    _Float16* Kb  = (_Float16*)(ws + 4292608);
    _Float16* Vt  = (_Float16*)(ws + 8486912);
    float*    Op  = (float*)(ws + 12681216);
    float*    Mst = (float*)(ws + 29458432);
    float*    Lst = (float*)(ws + 29589504);

    wconv_kernel<<<dim3((3 * CH * CH + 255) / 256), dim3(256), 0, stream>>>(Wq, Wk, Wv, Wt);
    proj_kernel<<<dim3(MT / 16, 24), dim3(32), 0, stream>>>(x, Wt, bq, bk, bv, Qb, Kb, Vt);
    attn_kernel<<<dim3(NTOK / 16 / 4, NSEG, NB), dim3(128), 0, stream>>>(Qb, Kb, Vt, Op, Mst, Lst);
    merge_kernel<<<dim3((MT * CH) / 256), dim3(256), 0, stream>>>(x, Op, Mst, Lst, out);
}
